// GAT_30657476559417
// MI455X (gfx1250) — matrix operand order repaired and checked by reading, not compiled, not run on hardware
//
#include <hip/hip_runtime.h>

// ---------------- problem constants (match reference) ----------------
#define N_NODES   50000
#define N_EDGES   800000
#define DIM       128
#define NUM_GRAPHS 512
#define E_TOTAL   (N_EDGES + N_NODES)   // edges + self loops

__constant__ const float NEG_ATT = 0.2f;
__constant__ const float NEG_ACT = 0.01f;
__constant__ const float EPSF    = 1e-16f;

typedef __attribute__((ext_vector_type(16))) _Float16 v16h;
typedef __attribute__((ext_vector_type(8)))  float    v8f;

// ---------------- helpers ----------------
__device__ __forceinline__ float lrelu(float x, float sl) { return x > 0.0f ? x : sl * x; }

// order-preserving float <-> uint mapping for atomicMax on floats
__device__ __forceinline__ unsigned fmap(float f) {
    unsigned u = __float_as_uint(f);
    return (u & 0x80000000u) ? ~u : (u | 0x80000000u);
}
__device__ __forceinline__ float funmap(unsigned u) {
    return (u & 0x80000000u) ? __uint_as_float(u & 0x7FFFFFFFu) : __uint_as_float(~u);
}

__device__ __forceinline__ void edge_sd(const int* __restrict__ ei, int idx, int& s, int& d) {
    if (idx < N_EDGES) { s = ei[idx]; d = ei[N_EDGES + idx]; }
    else               { s = idx - N_EDGES; d = s; }          // self loop
}

// ---------------- f32 -> f16 convert (optionally fold leaky_relu 0.01) ----------------
__global__ void cvt_to_f16(const float* __restrict__ in, _Float16* __restrict__ out,
                           int n, int apply_act) {
    int i = blockIdx.x * blockDim.x + threadIdx.x;
    if (i >= n) return;
    float v = in[i];
    if (apply_act) v = lrelu(v, NEG_ACT);
    out[i] = (_Float16)v;
}

// ---------------- WMMA GEMM: H[N,128] = Xh[N,128] @ Wh[128,128]^T ----------------
// One wave computes a 16x16 tile of H. Block = 256 threads = 8 waves covering
// one 16-row strip across all 8 column tiles. K=128 -> four 16x16x32 WMMAs.
__global__ __launch_bounds__(256) void gemm_xwT_wmma(const _Float16* __restrict__ Xh,
                                                     const _Float16* __restrict__ Wh,
                                                     float* __restrict__ H) {
    const int lane = threadIdx.x & 31;
    const int wave = threadIdx.x >> 5;
    const int m0   = blockIdx.x * 16;    // row tile (N_NODES = 3125*16 exactly)
    const int n0   = wave * 16;          // column tile (8 tiles cover DIM=128)
    const int g    = lane >> 4;          // lane group 0/1
    const int l16  = lane & 15;

    const _Float16* __restrict__ xrow = Xh + (size_t)(m0 + l16) * DIM; // A row M=l16
    const _Float16* __restrict__ wrow = Wh + (size_t)(n0 + l16) * DIM; // B col N=l16 (B=W^T)

    v8f c = {};
#pragma unroll
    for (int kk = 0; kk < DIM; kk += 32) {
        v16h a, b;
#pragma unroll
        for (int j = 0; j < 8; ++j) {
            // A 16x32 f16 layout: lanes 0-15 hold K 0..7 / 16..23 pairs, lanes 16-31 offset +8
            int ka = kk + ((j < 4) ? (2 * j) : (16 + 2 * (j - 4))) + 8 * g;
            a[2 * j]     = xrow[ka];
            a[2 * j + 1] = xrow[ka + 1];
            // B 32x16 f16 layout: same K order as A (K 0..7 / 16..23 pairs, lanes 16-31 offset +8)
            int kb = kk + ((j < 4) ? (2 * j) : (16 + 2 * (j - 4))) + 8 * g;
            b[2 * j]     = wrow[kb];
            b[2 * j + 1] = wrow[kb + 1];
        }
        c = __builtin_amdgcn_wmma_f32_16x16x32_f16(false, a, false, b, (short)0, c,
                                                   false, false);
    }
    // C/D 16x16 f32 layout: lane group g -> rows r + 8g, col = l16
#pragma unroll
    for (int r = 0; r < 8; ++r) {
        H[(size_t)(m0 + r + 8 * g) * DIM + n0 + l16] = c[r];
    }
}

// ---------------- attention logits: es = H . a_src, ed = H . a_dst (wave per node) ----
__global__ __launch_bounds__(256) void gemv_att(const float* __restrict__ H,
                                                const float* __restrict__ av_src,
                                                const float* __restrict__ av_dst,
                                                float* __restrict__ es,
                                                float* __restrict__ ed) {
    int node = blockIdx.x * 8 + (threadIdx.x >> 5);
    if (node >= N_NODES) return;
    int lane = threadIdx.x & 31;
    float4 h4 = ((const float4*)(H + (size_t)node * DIM))[lane];
    float4 s4 = ((const float4*)av_src)[lane];
    float4 d4 = ((const float4*)av_dst)[lane];
    float ss = h4.x * s4.x + h4.y * s4.y + h4.z * s4.z + h4.w * s4.w;
    float dd = h4.x * d4.x + h4.y * d4.y + h4.z * d4.z + h4.w * d4.w;
#pragma unroll
    for (int off = 16; off; off >>= 1) {
        ss += __shfl_xor(ss, off, 32);
        dd += __shfl_xor(dd, off, 32);
    }
    if (lane == 0) { es[node] = ss; ed[node] = dd; }
}

// ---------------- edge pass 1: segment max of attention logits ----------------
__global__ void edge_max(const int* __restrict__ ei, const float* __restrict__ es,
                         const float* __restrict__ ed, unsigned* __restrict__ mmax) {
    int idx = blockIdx.x * blockDim.x + threadIdx.x;
    if (idx >= E_TOTAL) return;
    int s, d; edge_sd(ei, idx, s, d);
    float e = lrelu(es[s] + ed[d], NEG_ATT);
    atomicMax(&mmax[d], fmap(e));
}

// ---------------- edge pass 2: segment sum of exp(e - m[dst]) ----------------
__global__ void edge_expsum(const int* __restrict__ ei, const float* __restrict__ es,
                            const float* __restrict__ ed, const unsigned* __restrict__ mmax,
                            float* __restrict__ z) {
    int idx = blockIdx.x * blockDim.x + threadIdx.x;
    if (idx >= E_TOTAL) return;
    int s, d; edge_sd(ei, idx, s, d);
    float e = lrelu(es[s] + ed[d], NEG_ATT);
    atomicAdd(&z[d], __expf(e - funmap(mmax[d])));
}

// ---------------- edge pass 3: agg[dst] += alpha * h[src]  (wave per edge) ---------
__global__ __launch_bounds__(256) void edge_aggregate(const int* __restrict__ ei,
                                                      const float* __restrict__ es,
                                                      const float* __restrict__ ed,
                                                      const unsigned* __restrict__ mmax,
                                                      const float* __restrict__ z,
                                                      const float* __restrict__ H,
                                                      float* __restrict__ agg) {
    int eidx = blockIdx.x * 8 + (threadIdx.x >> 5);
    if (eidx >= E_TOTAL) return;
    int lane = threadIdx.x & 31;
    int s, d; edge_sd(ei, eidx, s, d);
    float e     = lrelu(es[s] + ed[d], NEG_ATT);
    float alpha = __expf(e - funmap(mmax[d])) / (z[d] + EPSF);
    float4 h4 = ((const float4*)(H + (size_t)s * DIM))[lane];
    float* ag = agg + (size_t)d * DIM + lane * 4;
    atomicAdd(ag + 0, h4.x * alpha);
    atomicAdd(ag + 1, h4.y * alpha);
    atomicAdd(ag + 2, h4.z * alpha);
    atomicAdd(ag + 3, h4.w * alpha);
}

// ---------------- global_add_pool by batch (wave per node) ----------------
__global__ __launch_bounds__(256) void pool_batch(const float* __restrict__ agg,
                                                  const int* __restrict__ batch,
                                                  float* __restrict__ out) {
    int node = blockIdx.x * 8 + (threadIdx.x >> 5);
    if (node >= N_NODES) return;
    int lane = threadIdx.x & 31;
    int b = batch[node];
    float4 v = ((const float4*)(agg + (size_t)node * DIM))[lane];
    float* o = out + (size_t)b * DIM + lane * 4;
    atomicAdd(o + 0, v.x);
    atomicAdd(o + 1, v.y);
    atomicAdd(o + 2, v.z);
    atomicAdd(o + 3, v.w);
}

// ---------------- host-side layer driver ----------------
namespace {
struct Scratch {
    float*    h;      // N*DIM   f32 : h = lin(x)
    float*    agg;    // N*DIM   f32 : aggregated output (next layer's input)
    _Float16* xh;     // N*DIM   f16 : GEMM A operand
    _Float16* wh;     // DIM*DIM f16 : GEMM B operand (W row-major)
    float*    es;     // N       f32
    float*    ed;     // N       f32
    unsigned* mmax;   // N       u32 (mapped float max)
    float*    z;      // N       f32
};

inline size_t alignup(size_t v) { return (v + 255) & ~(size_t)255; }

void run_layer(const float* xin, int apply_act, const float* W,
               const float* a_src, const float* a_dst, Scratch& S, hipStream_t stream) {
    const int ND = N_NODES * DIM;
    // 1) convert inputs to f16 (fold previous activation if requested)
    cvt_to_f16<<<(ND + 255) / 256, 256, 0, stream>>>(xin, S.xh, ND, apply_act);
    cvt_to_f16<<<(DIM * DIM + 255) / 256, 256, 0, stream>>>(W, S.wh, DIM * DIM, 0);
    // 2) reset accumulators
    hipMemsetAsync(S.mmax, 0, N_NODES * sizeof(unsigned), stream); // maps to -max float
    hipMemsetAsync(S.z,    0, N_NODES * sizeof(float),    stream);
    hipMemsetAsync(S.agg,  0, (size_t)ND * sizeof(float), stream);
    // 3) h = x @ W^T via WMMA
    gemm_xwT_wmma<<<N_NODES / 16, 256, 0, stream>>>(S.xh, S.wh, S.h);
    // 4) attention logits
    gemv_att<<<(N_NODES + 7) / 8, 256, 0, stream>>>(S.h, a_src, a_dst, S.es, S.ed);
    // 5) segment softmax + weighted aggregation
    edge_max<<<(E_TOTAL + 255) / 256, 256, 0, stream>>>(nullptr, S.es, S.ed, S.mmax); // placeholder, fixed below
}
} // namespace

extern "C" void kernel_launch(void* const* d_in, const int* in_sizes, int n_in,
                              void* d_out, int out_size, void* d_ws, size_t ws_size,
                              hipStream_t stream) {
    (void)in_sizes; (void)n_in; (void)out_size; (void)ws_size;
    const float* x     = (const float*)d_in[0];
    const int*   ei    = (const int*)d_in[1];
    const int*   batch = (const int*)d_in[2];
    const float* W[3]     = {(const float*)d_in[3], (const float*)d_in[6], (const float*)d_in[9]};
    const float* a_src[3] = {(const float*)d_in[4], (const float*)d_in[7], (const float*)d_in[10]};
    const float* a_dst[3] = {(const float*)d_in[5], (const float*)d_in[8], (const float*)d_in[11]};
    float* out = (float*)d_out;

    const size_t ND = (size_t)N_NODES * DIM;
    char* p = (char*)d_ws;
    Scratch S;
    S.h    = (float*)p;                 p += alignup(ND * 4);
    S.agg  = (float*)p;                 p += alignup(ND * 4);
    S.xh   = (_Float16*)p;              p += alignup(ND * 2);
    S.wh   = (_Float16*)p;              p += alignup((size_t)DIM * DIM * 2);
    S.es   = (float*)p;                 p += alignup((size_t)N_NODES * 4);
    S.ed   = (float*)p;                 p += alignup((size_t)N_NODES * 4);
    S.mmax = (unsigned*)p;              p += alignup((size_t)N_NODES * 4);
    S.z    = (float*)p;                 p += alignup((size_t)N_NODES * 4);

    const int ndElems = N_NODES * DIM;
    const int edgeBlocks1 = (E_TOTAL + 255) / 256;
    const int edgeBlocksW = (E_TOTAL + 7) / 8;
    const int nodeBlocksW = (N_NODES + 7) / 8;

    const float* xin = x;
    for (int layer = 0; layer < 3; ++layer) {
        int act = (layer == 2) ? 1 : 0;  // leaky_relu(0.01) applied between conv2 and conv3
        cvt_to_f16<<<(ndElems + 255) / 256, 256, 0, stream>>>(xin, S.xh, ndElems, act);
        cvt_to_f16<<<(DIM * DIM + 255) / 256, 256, 0, stream>>>(W[layer], S.wh, DIM * DIM, 0);
        hipMemsetAsync(S.mmax, 0, N_NODES * sizeof(unsigned), stream);
        hipMemsetAsync(S.z,    0, N_NODES * sizeof(float),    stream);
        hipMemsetAsync(S.agg,  0, ND * sizeof(float),         stream);

        gemm_xwT_wmma<<<N_NODES / 16, 256, 0, stream>>>(S.xh, S.wh, S.h);
        gemv_att<<<nodeBlocksW, 256, 0, stream>>>(S.h, a_src[layer], a_dst[layer], S.es, S.ed);

        edge_max    <<<edgeBlocks1, 256, 0, stream>>>(ei, S.es, S.ed, S.mmax);
        edge_expsum <<<edgeBlocks1, 256, 0, stream>>>(ei, S.es, S.ed, S.mmax, S.z);
        edge_aggregate<<<edgeBlocksW, 256, 0, stream>>>(ei, S.es, S.ed, S.mmax, S.z, S.h, S.agg);

        xin = S.agg;  // next layer consumes aggregated features (converted at loop top)
    }

    hipMemsetAsync(out, 0, (size_t)NUM_GRAPHS * DIM * sizeof(float), stream);
    pool_batch<<<nodeBlocksW, 256, 0, stream>>>(S.agg, batch, out);
}
